// GridTransformerBlock_66898410602636
// MI455X (gfx1250) — matrix operand order repaired and checked by reading, not compiled, not run on hardware
//
#include <hip/hip_runtime.h>
#include <hip/hip_bf16.h>

// ---------------------------------------------------------------------------
// GridTransformerBlock for MI455X (gfx1250, wave32, WMMA bf16 16x16x32).
// One workgroup = one (block g, pair-of-patch-columns) slice: M = 32 tokens.
// GEMMs via v_wmma_f32_16x16x32_bf16 (fp32 accumulate); softmax / layernorm /
// residuals in fp32.  x slice staged fp32 in LDS via global_load_async_to_lds.
// ---------------------------------------------------------------------------

typedef __bf16 bf16_t;
typedef bf16_t v8bf  __attribute__((ext_vector_type(8)));
typedef bf16_t v16bf __attribute__((ext_vector_type(16)));
typedef float  v8f   __attribute__((ext_vector_type(8)));
typedef float  v4f   __attribute__((ext_vector_type(4)));

#define EPS_LN 1e-5f

// ---- LDS arena (byte offsets, all 16B aligned) ----------------------------
// xbf   [32][264] bf16 @      0 (16896) : x (bf16), later x1 (bf16) A-operand
// x32   [32][264] f32  @  16896 (33792) : fp32 x slice (async copy; residual)
// qkvbf [32][776] bf16 @  50688 (49664) : q|k|v ; overlay x1_32 f32 [32][264]
// obf   [32][264] bf16 @ 100352 (16896) : attn out ; overlay hbf [32][72]
// wstg  (36864 B)      @ 117248         : staged weights (bf16)
// red   [256] float2   @ 154112 ( 2048) : layernorm partial sums
#define OFF_XBF   0
#define OFF_X32   16896
#define OFF_QKV   50688
#define OFF_OBF   100352
#define OFF_WSTG  117248
#define OFF_RED   154112
#define SMEM_TOTAL 156160

#define SX  264   // stride of [32][264] buffers (elements)
#define SQ  776   // stride of qkvbf
#define SH  72    // stride of hbf / W2 staging (64 + 8 pad)

__device__ __forceinline__ v8f wmma_bf16(v16bf a, v16bf b, v8f c) {
    // (neg_a, A, neg_b, B, c_mod, C, reuse_a, reuse_b)
    return __builtin_amdgcn_wmma_f32_16x16x32_bf16(false, a, false, b,
                                                   (short)0, c, false, false);
}

// A fragment (16x32 bf16): lane l<16 -> row base+(l&15), K = {0..7, 16..23};
// lane l>=16 -> same rows, K = {8..15, 24..31}.  Row-major [row][K] source.
__device__ __forceinline__ v16bf load_frag_a(const bf16_t* buf, int row_base,
                                             int k0, int stride) {
    int lane = threadIdx.x & 31;
    int half = lane >> 4;
    const bf16_t* p = buf + (row_base + (lane & 15)) * stride + k0 + half * 8;
    v8bf lo = *(const v8bf*)p;
    v8bf hi = *(const v8bf*)(p + 16);
    return __builtin_shufflevector(lo, hi, 0,1,2,3,4,5,6,7,8,9,10,11,12,13,14,15);
}

// B fragment (32x16 bf16): lane l<16 -> column base+(l&15), K = {0..7, 16..23};
// lane l>=16 -> K = {8..15, 24..31}.  Source row-major [col][K] (i.e. weight rows).
__device__ __forceinline__ v16bf load_frag_b(const bf16_t* buf, int col_base,
                                             int k0, int stride) {
    int lane = threadIdx.x & 31;
    int half = lane >> 4;
    const bf16_t* p = buf + (col_base + (lane & 15)) * stride + k0 + half * 8;
    v8bf lo = *(const v8bf*)p;
    v8bf hi = *(const v8bf*)(p + 16);
    return __builtin_shufflevector(lo, hi, 0,1,2,3,4,5,6,7,8,9,10,11,12,13,14,15);
}

__global__ __launch_bounds__(256, 1)
void grid_tblock_kernel(const float* __restrict__ x,
                        const float* __restrict__ Wqkv, const float* __restrict__ bqkv,
                        const float* __restrict__ Wo,   const float* __restrict__ bo,
                        const float* __restrict__ ln1g, const float* __restrict__ ln1b,
                        const float* __restrict__ ln2g, const float* __restrict__ ln2b,
                        const float* __restrict__ W1,   const float* __restrict__ b1,
                        const float* __restrict__ W2,   const float* __restrict__ b2,
                        float* __restrict__ out) {
    __shared__ __align__(16) char smem[SMEM_TOTAL];
    bf16_t* xbf   = (bf16_t*)(smem + OFF_XBF);
    float*  x32   = (float*) (smem + OFF_X32);
    bf16_t* qkvbf = (bf16_t*)(smem + OFF_QKV);
    float*  x1_32 = (float*) (smem + OFF_QKV);   // overlay (after attention)
    bf16_t* obf   = (bf16_t*)(smem + OFF_OBF);
    bf16_t* hbf   = (bf16_t*)(smem + OFF_OBF);   // overlay (FF phase)
    bf16_t* wstg  = (bf16_t*)(smem + OFF_WSTG);
    float2* red   = (float2*)(smem + OFF_RED);

    const int tid  = threadIdx.x;
    const int lane = tid & 31;
    const int wave = tid >> 5;          // 0..7
    const int half = lane >> 4;
    const int colL = lane & 15;

    const int g  = blockIdx.y;          // transformer block 0..255
    const int n0 = blockIdx.x * 2;      // first of 2 patch columns
    const int bi = g >> 4, bj = g & 15;

    // global element index of (s,n,e) of this patch block
    auto gidx = [&](int s, int n, int e) -> size_t {
        return ((size_t)((bi * 16 + s) * 256 + (bj * 16 + n))) * 256 + e;
    };
    // local token row r (0..31)  <->  (s = r>>1, nn = r&1), n = n0+nn

    // ---- Phase 0a: async DMA x slice (fp32) -> LDS x32 --------------------
    {
        int r = tid >> 3;                // 0..31
        int e0 = (tid & 7) * 32;
        int s = r >> 1, nn = r & 1;
        unsigned voff  = (unsigned)(gidx(s, n0 + nn, e0) * 4u); // byte offset
        unsigned lds0  = (unsigned)(uintptr_t)(x32 + r * SX + e0);
        #pragma unroll
        for (int j = 0; j < 8; ++j) {     // 8 x 16B per lane
            unsigned ldsa = lds0 + j * 16u;
            unsigned va   = voff + j * 16u;
            asm volatile("global_load_async_to_lds_b128 %0, %1, %2"
                         :: "v"(ldsa), "v"(va), "s"(x) : "memory");
        }
        asm volatile("s_wait_asynccnt 0" ::: "memory");
    }
    __syncthreads();

    // ---- Phase 0b: convert x32 -> bf16 xbf --------------------------------
    {
        int r = tid >> 3;
        int e0 = (tid & 7) * 32;
        const float* src = x32 + r * SX + e0;
        bf16_t* dst = xbf + r * SX + e0;
        #pragma unroll
        for (int j = 0; j < 32; j += 4) {
            v4f v = *(const v4f*)(src + j);
            dst[j + 0] = (bf16_t)v.x; dst[j + 1] = (bf16_t)v.y;
            dst[j + 2] = (bf16_t)v.z; dst[j + 3] = (bf16_t)v.w;
        }
    }
    __syncthreads();

    // ---- Phase 1: QKV GEMM  (M=32, K=256, N=768 in chunks of 64) ----------
    const int mt = wave & 1;             // 2 M-tiles
    const int nt = wave >> 1;            // 4 N-tiles per 64-wide chunk
    for (int nc = 0; nc < 12; ++nc) {
        { // stage 64 weight rows (fp32 -> bf16)
            int wr = tid >> 2;           // 0..63
            int e0 = (tid & 3) * 64;
            const float* src = Wqkv + (size_t)g * 768 * 256
                                    + (size_t)(nc * 64 + wr) * 256 + e0;
            if (nc + 1 < 12) __builtin_prefetch(src + 64 * 256, 0, 0);
            bf16_t* dst = wstg + wr * SX + e0;
            #pragma unroll
            for (int j = 0; j < 64; j += 4) {
                v4f v = *(const v4f*)(src + j);
                dst[j + 0] = (bf16_t)v.x; dst[j + 1] = (bf16_t)v.y;
                dst[j + 2] = (bf16_t)v.z; dst[j + 3] = (bf16_t)v.w;
            }
        }
        __syncthreads();
        v8f acc = {};
        #pragma unroll
        for (int k0 = 0; k0 < 256; k0 += 32) {
            v16bf a = load_frag_a(xbf,  mt * 16, k0, SX);
            v16bf b = load_frag_b(wstg, nt * 16, k0, SX);
            acc = wmma_bf16(a, b, acc);
        }
        int colF = nc * 64 + nt * 16 + colL;
        float bias = bqkv[(size_t)g * 768 + colF];
        #pragma unroll
        for (int v = 0; v < 8; ++v) {
            int row = mt * 16 + v + 8 * half;
            qkvbf[row * SQ + colF] = (bf16_t)(acc[v] + bias);
        }
        __syncthreads();
    }

    // ---- Phase 2: attention (VALU; per (nn, head, s) thread) --------------
    {
        int nn = tid >> 7;               // 0..1
        int rem = tid & 127;
        int h = rem >> 4, s = rem & 15;
        int rq = s * 2 + nn;
        float q[32];
        {
            const v8bf* q8 = (const v8bf*)(qkvbf + rq * SQ + h * 32);
            #pragma unroll
            for (int c = 0; c < 4; ++c) {
                v8bf t = q8[c];
                #pragma unroll
                for (int i = 0; i < 8; ++i) q[c * 8 + i] = (float)t[i];
            }
        }
        float sc[16]; float mx = -1e30f;
        #pragma unroll
        for (int t = 0; t < 16; ++t) {
            const v8bf* k8 = (const v8bf*)(qkvbf + (t * 2 + nn) * SQ + 256 + h * 32);
            float acc = 0.f;
            #pragma unroll
            for (int c = 0; c < 4; ++c) {
                v8bf kv = k8[c];
                #pragma unroll
                for (int i = 0; i < 8; ++i) acc += q[c * 8 + i] * (float)kv[i];
            }
            acc *= 0.17677669529663687f;         // 1/sqrt(32)
            sc[t] = acc; mx = fmaxf(mx, acc);
        }
        float sum = 0.f;
        #pragma unroll
        for (int t = 0; t < 16; ++t) { float p = __expf(sc[t] - mx); sc[t] = p; sum += p; }
        float inv = 1.f / sum;
        float o[32];
        #pragma unroll
        for (int d = 0; d < 32; ++d) o[d] = 0.f;
        #pragma unroll
        for (int t = 0; t < 16; ++t) {
            float w = sc[t] * inv;
            const v8bf* v8p = (const v8bf*)(qkvbf + (t * 2 + nn) * SQ + 512 + h * 32);
            #pragma unroll
            for (int c = 0; c < 4; ++c) {
                v8bf vv = v8p[c];
                #pragma unroll
                for (int i = 0; i < 8; ++i) o[c * 8 + i] += w * (float)vv[i];
            }
        }
        bf16_t* orow = obf + rq * SX + h * 32;
        #pragma unroll
        for (int d = 0; d < 32; ++d) orow[d] = (bf16_t)o[d];
    }
    __syncthreads();

    // ---- Phase 3: O-projection + residual (fp32 from LDS x32) -------------
    for (int nc = 0; nc < 4; ++nc) {
        {
            int wr = tid >> 2;
            int e0 = (tid & 3) * 64;
            const float* src = Wo + (size_t)g * 256 * 256
                                  + (size_t)(nc * 64 + wr) * 256 + e0;
            if (nc + 1 < 4) __builtin_prefetch(src + 64 * 256, 0, 0);
            bf16_t* dst = wstg + wr * SX + e0;
            #pragma unroll
            for (int j = 0; j < 64; j += 4) {
                v4f v = *(const v4f*)(src + j);
                dst[j + 0] = (bf16_t)v.x; dst[j + 1] = (bf16_t)v.y;
                dst[j + 2] = (bf16_t)v.z; dst[j + 3] = (bf16_t)v.w;
            }
        }
        __syncthreads();
        v8f acc = {};
        #pragma unroll
        for (int k0 = 0; k0 < 256; k0 += 32) {
            v16bf a = load_frag_a(obf,  mt * 16, k0, SX);
            v16bf b = load_frag_b(wstg, nt * 16, k0, SX);
            acc = wmma_bf16(a, b, acc);
        }
        int colE = nc * 64 + nt * 16 + colL;
        float bias = bo[(size_t)g * 256 + colE];
        #pragma unroll
        for (int v = 0; v < 8; ++v) {
            int row = mt * 16 + v + 8 * half;
            float xv = x32[row * SX + colE];             // fp32 residual (LDS)
            x1_32[row * SX + colE] = acc[v] + bias + xv;
        }
        __syncthreads();
    }

    // ---- Phase 4: LayerNorm 1 (fp32 in place, bf16 copy into xbf) ---------
    {
        int r = tid >> 3; int j = tid & 7; int e0 = j * 32;
        const float* p = x1_32 + r * SX + e0;
        float sm = 0.f, sq = 0.f;
        #pragma unroll
        for (int k = 0; k < 32; ++k) { float v = p[k]; sm += v; sq += v * v; }
        red[(r << 3) + j] = make_float2(sm, sq);
        __syncthreads();
        float tsm = 0.f, tsq = 0.f;
        #pragma unroll
        for (int k = 0; k < 8; ++k) { float2 t = red[(r << 3) + k]; tsm += t.x; tsq += t.y; }
        float m   = tsm * (1.f / 256.f);
        float var = tsq * (1.f / 256.f) - m * m;
        float rs  = rsqrtf(var + EPS_LN);
        float*  pr = x1_32 + r * SX + e0;
        bf16_t* br = xbf   + r * SX + e0;
        const float* gg = ln1g + (size_t)g * 256 + e0;
        const float* bb = ln1b + (size_t)g * 256 + e0;
        #pragma unroll
        for (int k = 0; k < 32; ++k) {
            float v = (pr[k] - m) * rs * gg[k] + bb[k];
            pr[k] = v; br[k] = (bf16_t)v;
        }
    }
    __syncthreads();

    // ---- Phase 5: FFN (F in chunks of 64; FF2 acc in WMMA C fragments) ----
    v8f facc[4];
    #pragma unroll
    for (int i = 0; i < 4; ++i) { v8f z = {}; facc[i] = z; }
    for (int fc = 0; fc < 16; ++fc) {
        { // stage W1 rows [fc*64 .. +64)
            int wr = tid >> 2;           // 0..63
            int e0 = (tid & 3) * 64;
            const float* src = W1 + (size_t)g * 1024 * 256
                                  + (size_t)(fc * 64 + wr) * 256 + e0;
            if (fc + 1 < 16) __builtin_prefetch(src + 64 * 256, 0, 0);
            bf16_t* dst = wstg + wr * SX + e0;
            #pragma unroll
            for (int j = 0; j < 64; j += 4) {
                v4f v = *(const v4f*)(src + j);
                dst[j + 0] = (bf16_t)v.x; dst[j + 1] = (bf16_t)v.y;
                dst[j + 2] = (bf16_t)v.z; dst[j + 3] = (bf16_t)v.w;
            }
        }
        __syncthreads();
        { // FF1: h-chunk [32][64] = relu(x1 @ W1^T + b1)
            v8f c0 = {};
            #pragma unroll
            for (int k0 = 0; k0 < 256; k0 += 32) {
                v16bf a = load_frag_a(xbf,  mt * 16, k0, SX);
                v16bf b = load_frag_b(wstg, nt * 16, k0, SX);
                c0 = wmma_bf16(a, b, c0);
            }
            int cl = nt * 16 + colL;
            float bias = b1[(size_t)g * 1024 + fc * 64 + cl];
            #pragma unroll
            for (int v = 0; v < 8; ++v) {
                int row = mt * 16 + v + 8 * half;
                hbf[row * SH + cl] = (bf16_t)fmaxf(c0[v] + bias, 0.f);
            }
        }
        __syncthreads();
        { // stage W2 columns [fc*64 .. +64) for all 256 output rows e
            int er = tid;                // 0..255
            const float* src = W2 + (size_t)g * 256 * 1024
                                  + (size_t)er * 1024 + fc * 64;
            if (fc + 1 < 16) __builtin_prefetch(src + 64, 0, 0);
            bf16_t* dst = wstg + er * SH;
            #pragma unroll
            for (int j = 0; j < 64; j += 4) {
                v4f v = *(const v4f*)(src + j);
                dst[j + 0] = (bf16_t)v.x; dst[j + 1] = (bf16_t)v.y;
                dst[j + 2] = (bf16_t)v.z; dst[j + 3] = (bf16_t)v.w;
            }
        }
        __syncthreads();
        { // FF2 partial: facc += h_chunk @ W2_chunk^T
            #pragma unroll
            for (int k0 = 0; k0 < 64; k0 += 32) {
                v16bf a = load_frag_a(hbf, mt * 16, k0, SH);
                #pragma unroll
                for (int i = 0; i < 4; ++i) {
                    v16bf b = load_frag_b(wstg, (nt + 4 * i) * 16, k0, SH);
                    facc[i] = wmma_bf16(a, b, facc[i]);
                }
            }
        }
        __syncthreads();
    }
    // FF epilogue: + b2 + residual(x1 fp32), write back into x1_32
    #pragma unroll
    for (int i = 0; i < 4; ++i) {
        int colE = (nt + 4 * i) * 16 + colL;
        float bias = b2[(size_t)g * 256 + colE];
        #pragma unroll
        for (int v = 0; v < 8; ++v) {
            int row = mt * 16 + v + 8 * half;
            x1_32[row * SX + colE] = facc[i][v] + bias + x1_32[row * SX + colE];
        }
    }
    __syncthreads();

    // ---- Phase 6: LayerNorm 2 -> global output ----------------------------
    {
        int r = tid >> 3; int j = tid & 7; int e0 = j * 32;
        const float* p = x1_32 + r * SX + e0;
        float sm = 0.f, sq = 0.f;
        #pragma unroll
        for (int k = 0; k < 32; ++k) { float v = p[k]; sm += v; sq += v * v; }
        red[(r << 3) + j] = make_float2(sm, sq);
        __syncthreads();
        float tsm = 0.f, tsq = 0.f;
        #pragma unroll
        for (int k = 0; k < 8; ++k) { float2 t = red[(r << 3) + k]; tsm += t.x; tsq += t.y; }
        float m   = tsm * (1.f / 256.f);
        float var = tsq * (1.f / 256.f) - m * m;
        float rs  = rsqrtf(var + EPS_LN);
        int s = r >> 1, nn = r & 1;
        float* dst = out + gidx(s, n0 + nn, e0);
        const float* gg = ln2g + (size_t)g * 256 + e0;
        const float* bb = ln2b + (size_t)g * 256 + e0;
        #pragma unroll
        for (int k = 0; k < 32; ++k)
            dst[k] = (p[k] - m) * rs * gg[k] + bb[k];
    }
}

extern "C" void kernel_launch(void* const* d_in, const int* in_sizes, int n_in,
                              void* d_out, int out_size, void* d_ws, size_t ws_size,
                              hipStream_t stream) {
    const float* x    = (const float*)d_in[0];
    const float* Wqkv = (const float*)d_in[1];
    const float* bqkv = (const float*)d_in[2];
    const float* Wo   = (const float*)d_in[3];
    const float* bo   = (const float*)d_in[4];
    const float* l1g  = (const float*)d_in[5];
    const float* l1b  = (const float*)d_in[6];
    const float* l2g  = (const float*)d_in[7];
    const float* l2b  = (const float*)d_in[8];
    const float* W1   = (const float*)d_in[9];
    const float* b1   = (const float*)d_in[10];
    const float* W2   = (const float*)d_in[11];
    const float* b2   = (const float*)d_in[12];
    dim3 grid(8, 256, 1);   // x = n-chunk (fast) so chunks of one block share L2 weight lines
    hipLaunchKernelGGL(grid_tblock_kernel, grid, dim3(256, 1, 1), 0, stream,
                       x, Wqkv, bqkv, Wo, bo, l1g, l1b, l2g, l2b,
                       W1, b1, W2, b2, (float*)d_out);
}
